// SentenceDecoder_88201448391249
// MI455X (gfx1250) — matrix operand order repaired and checked by reading, not compiled, not run on hardware
//
#include <hip/hip_runtime.h>
#include <hip/hip_bf16.h>
#include <cstddef>

// Problem constants
#define Bn 32
#define Pn 49
#define En 256
#define Hn 256
#define Vn 32000
#define Tn 48
#define STEPS 47
#define KIN 1056          // padded 792+256 -> 33 K-blocks of 32
#define KBIN 33

typedef __attribute__((ext_vector_type(16))) __bf16 v16bf;
typedef __attribute__((ext_vector_type(8)))  float  v8f;

union ABfrag { v16bf v; unsigned int u[8]; };

// native f32->bf16 (backend emits hardware cvt, RNE)
__device__ __forceinline__ unsigned int pack2(float a, float b) {
    union { __bf16 h[2]; unsigned int u; } x;
    x.h[0] = (__bf16)a;
    x.h[1] = (__bf16)b;
    return x.u;
}

__device__ __forceinline__ v8f wmma_bf16(v16bf a, v16bf b, v8f c) {
    return __builtin_amdgcn_wmma_f32_16x16x32_bf16(false, a, false, b, (short)0, c, false, false);
}

// Load a pre-packed fragment (A or B): layout [tile][kb][lane][8 u32] contiguous.
__device__ __forceinline__ v16bf load_frag(const unsigned int* pk, int tile, int kb, int KB) {
    int lane = threadIdx.x & 31;
    const unsigned int* p = pk + (((size_t)tile * KB + kb) * 32 + lane) * 8;
    ABfrag f;
#pragma unroll
    for (int j = 0; j < 8; ++j) f.u[j] = p[j];
    return f.v;
}

// ---------------- A-fragment packing (activations, per step) ----------------
// A 16x32 bf16 tile: lane=M(mod16), k0 = kb*32 + (j>=4?16:0) + (lane>=16?8:0) + (j&3)*2
__global__ void k_packa(const float* src, int ld, int MT, int KB, unsigned int* dst) {
    int total = MT * KB * 256;   // * 32 lanes * 8 u32
    for (int i = blockIdx.x * blockDim.x + threadIdx.x; i < total;
         i += gridDim.x * blockDim.x) {
        int j = i & 7;
        int r = i >> 3;
        int lane = r & 31; r >>= 5;
        int kb = r % KB;
        int mt = r / KB;
        int m = mt * 16 + (lane & 15);
        int k0 = kb * 32 + ((j >> 2) << 4) + ((lane >> 4) << 3) + ((j & 3) << 1);
        const float* p = src + (size_t)m * ld + k0;
        dst[i] = pack2(p[0], p[1]);
    }
}

// ---------------- B-tile weight packing (once per launch) ----------------
// B 32x16 bf16 tile: lane=N(mod16), k0 = kb*32 + (j>=4?16:0) + (lane>=16?8:0) + (j&3)*2
// mode 0: Wp    (32000x256)
// mode 1: Wcat  (1024x1056)  = [Wih(1024x792) | Whh(1024x256) | 0 pad]
// mode 2: Wfc   (512x256)
// mode 3: W2cat (512x512)    = blockdiag(Whh2(256x256), Ws(256x256))
// mode 4: Wv    (256x256)
__device__ __forceinline__ float fetch_w(int mode, const float* s0, const float* s1, int n, int k) {
    switch (mode) {
        case 0: return s0[(size_t)n * 256 + k];
        case 1:
            if (k < 792)  return s0[(size_t)n * 792 + k];
            if (k < 1048) return s1[(size_t)n * 256 + (k - 792)];
            return 0.f;
        case 2: return s0[(size_t)n * 256 + k];
        case 3:
            if (n < 256 && k < 256)   return s0[(size_t)n * 256 + k];
            if (n >= 256 && k >= 256) return s1[(size_t)(n - 256) * 256 + (k - 256)];
            return 0.f;
        default: return s0[(size_t)n * 256 + k];
    }
}

__global__ void k_pack(const float* s0, const float* s1, unsigned int* dst,
                       int NT, int KB, int mode) {
    long total = (long)NT * KB * 32 * 8;
    for (long i = (long)blockIdx.x * blockDim.x + threadIdx.x; i < total;
         i += (long)gridDim.x * blockDim.x) {
        int j = (int)(i & 7);
        long r = i >> 3;
        int lane = (int)(r & 31); r >>= 5;
        int kb = (int)(r % KB);
        int nt = (int)(r / KB);
        int n = nt * 16 + (lane & 15);
        int k0 = kb * 32 + ((j >> 2) << 4) + ((lane >> 4) << 3) + ((j & 3) << 1);
        dst[i] = pack2(fetch_w(mode, s0, s1, n, k0),
                       fetch_w(mode, s0, s1, n, k0 + 1));
    }
}

// ---------------- setup: image_mean, x_static, h0, m0 ----------------
__global__ void k_init(const float* image, const float* vp, const float* label,
                       const float* topic, const float* Wh, const float* bh,
                       const float* Wm, const float* bm,
                       float* xfull, float* h, float* m) {
    __shared__ float mean[En];
    int b = blockIdx.x, e = threadIdx.x;
    float s = 0.f;
    for (int p = 0; p < Pn; ++p) s += image[((size_t)b * Pn + p) * En + e];
    s *= (1.f / 49.f);
    mean[e] = s;
    float* xb = xfull + (size_t)b * KIN;
    xb[e] = s;                                  // [0,256)  image_mean
    if (e < 8)  xb[256 + e] = vp[b * 8 + e];    // [256,264) view_position
    if (e < 16) xb[264 + e] = label[b * 16 + e];// [264,280) label
    xb[280 + e] = topic[(size_t)b * Hn + e];    // [280,536) topic
    if (e < 8)  xb[1048 + e] = 0.f;             // pad
    __syncthreads();
    float ah = bh[e], am = bm[e];
    for (int k = 0; k < En; ++k) {
        float mk = mean[k];
        ah += mk * Wh[(size_t)e * En + k];
        am += mk * Wm[(size_t)e * En + k];
    }
    h[(size_t)b * Hn + e] = tanhf(ah);
    m[(size_t)b * Hn + e] = tanhf(am);
}

// ---------------- v = image @ Wv^T + bv  (M=1568, N=256, K=256) ----------------
__global__ void k_gemm_v(const unsigned int* pAimg, const unsigned int* pWv,
                         const float* bv, float* v) {
    int wave = (blockIdx.x * blockDim.x + threadIdx.x) >> 5;
    if (wave >= 98 * 16) return;
    int mt = wave >> 4, nt = wave & 15;
    v8f acc = {0, 0, 0, 0, 0, 0, 0, 0};
#pragma unroll
    for (int kb = 0; kb < 8; ++kb) {
        v16bf a = load_frag(pAimg, mt, kb, 8);
        v16bf b = load_frag(pWv, nt, kb, 8);
        acc = wmma_bf16(a, b, acc);
    }
    int lane = threadIdx.x & 31;
    int n = nt * 16 + (lane & 15);
    float bb = bv[n];
#pragma unroll
    for (int r = 0; r < 8; ++r) {
        int mrow = mt * 16 + r + ((lane >> 4) << 3);
        v[(size_t)mrow * En + n] = acc[r] + bb;
    }
}

// ---------------- generic M=32 GEMM: C(32xN) = A(32xK) @ Bpk^T + epilogue ----
// A is pre-packed (2 M-tiles). Epilogues:
// epi 0: + bias0[n] + bias1[n]                         (gates)
// epi 1: relu(+ bias0[n])                              (fc)
// epi 2: + (n<256 ? bias0[n] : bias1[n-256])           (block-diag proj)
// epi 3: (+ bias0[n]) / aux[m]                         (logits / temp)
__global__ void k_gemm32(const unsigned int* pA, const unsigned int* Bpk,
                         int NT, int KB, const float* bias0, const float* bias1,
                         const float* aux, float* C, int ldc, int epi) {
    int wave = (blockIdx.x * blockDim.x + threadIdx.x) >> 5;
    if (wave >= NT) return;
    int nt = wave;
    v8f acc0 = {0, 0, 0, 0, 0, 0, 0, 0};
    v8f acc1 = {0, 0, 0, 0, 0, 0, 0, 0};
    for (int kb = 0; kb < KB; ++kb) {
        v16bf b  = load_frag(Bpk, nt, kb, KB);
        v16bf a0 = load_frag(pA, 0, kb, KB);
        v16bf a1 = load_frag(pA, 1, kb, KB);
        acc0 = wmma_bf16(a0, b, acc0);
        acc1 = wmma_bf16(a1, b, acc1);
    }
    int lane = threadIdx.x & 31;
    int n = nt * 16 + (lane & 15);
    float bval;
    if (epi == 2) bval = (n < 256) ? bias0[n] : bias1[n - 256];
    else {
        bval = bias0 ? bias0[n] : 0.f;
        if (bias1) bval += bias1[n];
    }
#pragma unroll
    for (int r = 0; r < 8; ++r) {
        int m0 = r + ((lane >> 4) << 3);     // rows 0..15 (acc0), 16..31 (acc1)
        float x0 = acc0[r] + bval;
        float x1 = acc1[r] + bval;
        if (epi == 1) { x0 = fmaxf(x0, 0.f); x1 = fmaxf(x1, 0.f); }
        else if (epi == 3) { x0 /= aux[m0]; x1 /= aux[m0 + 16]; }
        C[(size_t)m0 * ldc + n]        = x0;
        C[(size_t)(m0 + 16) * ldc + n] = x1;
    }
}

// ---------------- per-step x packing: [536..792)=emb_t [792..1048)=h, pad 0 ---
__global__ void k_packx(float* xfull, const float* emb, const int* text,
                        const float* h, int t) {
    int i = blockIdx.x * blockDim.x + threadIdx.x;
    if (i >= Bn * 520) return;
    int b = i / 520, k = i % 520 + 536;
    float val;
    if (k < 792) {
        int tok = text[b * Tn + t];
        val = emb[(size_t)tok * En + (k - 536)];
    } else if (k < 1048) {
        val = h[(size_t)b * Hn + (k - 792)];
    } else val = 0.f;
    xfull[(size_t)b * KIN + k] = val;
}

// ---------------- LSTM pointwise ----------------
__global__ void k_lstm(const float* g, float* h, float* m) {
    int i = blockIdx.x * blockDim.x + threadIdx.x;
    if (i >= Bn * Hn) return;
    int b = i >> 8, j = i & 255;
    const float* gb = g + (size_t)b * 1024;
    float ig = 1.f / (1.f + __expf(-gb[j]));
    float fg = 1.f / (1.f + __expf(-gb[256 + j]));
    float gg = tanhf(gb[512 + j]);
    float og = 1.f / (1.f + __expf(-gb[768 + j]));
    float mm = fg * m[i] + ig * gg;
    m[i] = mm;
    h[i] = og * tanhf(mm);
}

// ---------------- attention + context ----------------
// hs2 = [_hh | _s] (Bx512), fo = [hh | s] (Bx512)
__global__ void k_attn(const float* v, const float* hs2, const float* fo,
                       const float* image, const float* Wz, const float* bz,
                       const int* length, float* att_out, float* xv, int t) {
    __shared__ float a_sh[50];
    __shared__ float zz[50];
    int b = blockIdx.x, tid = threadIdx.x;
    const float* _hh = hs2 + (size_t)b * 512;
    const float* _s  = _hh + 256;
    if (tid < 50) {
        float acc = bz[0];
        const float* vrow = (tid < 49) ? (v + ((size_t)b * Pn + tid) * En) : _s;
        for (int e = 0; e < En; ++e) acc += tanhf(vrow[e] + _hh[e]) * Wz[e];
        zz[tid] = acc;
    }
    __syncthreads();
    if (tid == 0) {
        float mx = -INFINITY;
        for (int p = 0; p < 50; ++p) mx = fmaxf(mx, zz[p]);
        float s = 0.f;
        for (int p = 0; p < 50; ++p) { float e = __expf(zz[p] - mx); a_sh[p] = e; s += e; }
        float inv = 1.f / s;
        for (int p = 0; p < 50; ++p) a_sh[p] *= inv;
    }
    __syncthreads();
    float valid = (t < length[b] - 1) ? 1.f : 0.f;
    if (tid < 50) att_out[((size_t)b * STEPS + t) * 50 + tid] = a_sh[tid] * valid;
    // context: c[e] = sum_p a[p]*img[p,e] + a[49]*s[e];  xv = c + hh
    int e = tid;
    const float* hh = fo + (size_t)b * 512;
    const float* sr = hh + 256;
    float c = a_sh[49] * sr[e];
    for (int p = 0; p < Pn; ++p) c += a_sh[p] * image[((size_t)b * Pn + p) * En + e];
    xv[(size_t)b * En + e] = c + hh[e];
}

// ---------------- row log-softmax (+argmax, masking) ----------------
__global__ void k_logsoftmax(const float* logits, const int* length,
                             float* logp_out, float* txt_out, int t) {
    __shared__ float smax[256];
    __shared__ int   sidx[256];
    __shared__ float ssum[256];
    int b = blockIdx.x, tid = threadIdx.x;
    const float* row = logits + (size_t)b * Vn;
    float mx = -INFINITY; int mi = 0;
    for (int n = tid; n < Vn; n += 256) {
        float x = row[n];
        if (x > mx) { mx = x; mi = n; }
    }
    smax[tid] = mx; sidx[tid] = mi;
    __syncthreads();
    for (int s = 128; s > 0; s >>= 1) {
        if (tid < s) {
            if (smax[tid + s] > smax[tid] ||
                (smax[tid + s] == smax[tid] && sidx[tid + s] < sidx[tid])) {
                smax[tid] = smax[tid + s]; sidx[tid] = sidx[tid + s];
            }
        }
        __syncthreads();
    }
    float rmax = smax[0]; int amax = sidx[0];
    float sum = 0.f;
    for (int n = tid; n < Vn; n += 256) sum += __expf(row[n] - rmax);
    ssum[tid] = sum;
    __syncthreads();
    for (int s = 128; s > 0; s >>= 1) {
        if (tid < s) ssum[tid] += ssum[tid + s];
        __syncthreads();
    }
    float lse = rmax + __logf(ssum[0]);
    float valid = (t < length[b] - 1) ? 1.f : 0.f;
    float* out = logp_out + ((size_t)b * STEPS + t) * Vn;
    for (int n = tid; n < Vn; n += 256) out[n] = (row[n] - lse) * valid;
    if (tid == 0) txt_out[b * STEPS + t] = (valid != 0.f) ? (float)amax : 0.f;
}

extern "C" void kernel_launch(void* const* d_in, const int* in_sizes, int n_in,
                              void* d_out, int out_size, void* d_ws, size_t ws_size,
                              hipStream_t stream) {
    (void)in_sizes; (void)n_in; (void)out_size; (void)ws_size;
    const float* image = (const float*)d_in[0];
    const float* vp    = (const float*)d_in[1];
    const float* label = (const float*)d_in[2];
    const float* topic = (const float*)d_in[3];
    const float* temp  = (const float*)d_in[4];
    const int*   text  = (const int*)d_in[5];
    const int*   length= (const int*)d_in[6];
    const float* emb   = (const float*)d_in[7];
    const float* Wv    = (const float*)d_in[8];  const float* bv   = (const float*)d_in[9];
    const float* Wh    = (const float*)d_in[10]; const float* bh   = (const float*)d_in[11];
    const float* Wm    = (const float*)d_in[12]; const float* bm   = (const float*)d_in[13];
    const float* Wih   = (const float*)d_in[14]; const float* bih  = (const float*)d_in[15];
    const float* Whh   = (const float*)d_in[16]; const float* bhh  = (const float*)d_in[17];
    const float* Wfc   = (const float*)d_in[18]; const float* bfc  = (const float*)d_in[19];
    const float* Whh2  = (const float*)d_in[20]; const float* bhh2 = (const float*)d_in[21];
    const float* Ws    = (const float*)d_in[22]; const float* bs   = (const float*)d_in[23];
    const float* Wz    = (const float*)d_in[24]; const float* bz   = (const float*)d_in[25];
    const float* Wp    = (const float*)d_in[26]; const float* bp   = (const float*)d_in[27];

    // workspace carve-up (256B aligned)
    char* w = (char*)d_ws;
    auto carve = [&](size_t bytes) -> void* {
        void* p = (void*)w;
        w += (bytes + 255) & ~(size_t)255;
        return p;
    };
    float* xfull  = (float*)carve((size_t)Bn * KIN * 4);
    float* h      = (float*)carve((size_t)Bn * Hn * 4);
    float* m      = (float*)carve((size_t)Bn * Hn * 4);
    float* vbuf   = (float*)carve((size_t)Bn * Pn * En * 4);
    float* g      = (float*)carve((size_t)Bn * 1024 * 4);
    float* fo     = (float*)carve((size_t)Bn * 512 * 4);
    float* hs2    = (float*)carve((size_t)Bn * 512 * 4);
    float* xv     = (float*)carve((size_t)Bn * En * 4);
    float* logits = (float*)carve((size_t)Bn * Vn * 4);
    // packed B tiles (weights)
    unsigned int* pWcat = (unsigned int*)carve((size_t)64 * KBIN * 256 * 4);
    unsigned int* pWfc  = (unsigned int*)carve((size_t)32 * 8 * 256 * 4);
    unsigned int* pW2   = (unsigned int*)carve((size_t)32 * 16 * 256 * 4);
    unsigned int* pWv   = (unsigned int*)carve((size_t)16 * 8 * 256 * 4);
    unsigned int* pWp   = (unsigned int*)carve((size_t)2000 * 8 * 256 * 4);
    // packed A fragments (activations)
    unsigned int* pAimg = (unsigned int*)carve((size_t)98 * 8 * 256 * 4);
    unsigned int* pAx   = (unsigned int*)carve((size_t)2 * KBIN * 256 * 4);
    unsigned int* pAh   = (unsigned int*)carve((size_t)2 * 8 * 256 * 4);
    unsigned int* pAfo  = (unsigned int*)carve((size_t)2 * 16 * 256 * 4);
    unsigned int* pAxv  = (unsigned int*)carve((size_t)2 * 8 * 256 * 4);

    // ---- setup ----
    k_init<<<Bn, 256, 0, stream>>>(image, vp, label, topic, Wh, bh, Wm, bm, xfull, h, m);
    k_pack<<<2112, 256, 0, stream>>>(Wih,  Whh,    pWcat, 64,   KBIN, 1);
    k_pack<<<256,  256, 0, stream>>>(Wfc,  nullptr,pWfc,  32,   8,    2);
    k_pack<<<512,  256, 0, stream>>>(Whh2, Ws,     pW2,   32,   16,   3);
    k_pack<<<128,  256, 0, stream>>>(Wv,   nullptr,pWv,   16,   8,    4);
    k_pack<<<16000,256, 0, stream>>>(Wp,   nullptr,pWp,   2000, 8,    0);
    k_packa<<<784, 256, 0, stream>>>(image, En, 98, 8, pAimg);
    k_gemm_v<<<392, 128, 0, stream>>>(pAimg, pWv, bv, vbuf);

    float* att_out  = (float*)d_out;
    float* logp_out = att_out + (size_t)Bn * STEPS * 50;
    float* txt_out  = logp_out + (size_t)Bn * STEPS * Vn;

    // ---- 47 sequential decode steps ----
    for (int t = 0; t < STEPS; ++t) {
        k_packx<<<(Bn * 520 + 255) / 256, 256, 0, stream>>>(xfull, emb, text, h, t);
        k_packa<<<66, 256, 0, stream>>>(xfull, KIN, 2, KBIN, pAx);
        // gates: (32x1024) = xfull(32x1056) @ Wcat^T + bih + bhh
        k_gemm32<<<16, 128, 0, stream>>>(pAx, pWcat, 64, KBIN, bih, bhh,
                                         nullptr, g, 1024, 0);
        k_lstm<<<Bn, 256, 0, stream>>>(g, h, m);
        k_packa<<<16, 256, 0, stream>>>(h, Hn, 2, 8, pAh);
        // fo = relu(h @ Wfc^T + bfc)
        k_gemm32<<<8, 128, 0, stream>>>(pAh, pWfc, 32, 8, bfc, nullptr,
                                        nullptr, fo, 512, 1);
        k_packa<<<32, 256, 0, stream>>>(fo, 512, 2, 16, pAfo);
        // [_hh|_s] = fo @ blockdiag(Whh2,Ws)^T + [bhh2|bs]
        k_gemm32<<<8, 128, 0, stream>>>(pAfo, pW2, 32, 16, bhh2, bs,
                                        nullptr, hs2, 512, 2);
        k_attn<<<Bn, 256, 0, stream>>>(vbuf, hs2, fo, image, Wz, bz, length,
                                       att_out, xv, t);
        k_packa<<<16, 256, 0, stream>>>(xv, En, 2, 8, pAxv);
        // logits/temp = (xv @ Wp^T + bp) / temp
        k_gemm32<<<500, 128, 0, stream>>>(pAxv, pWp, 2000, 8, bp, nullptr,
                                          temp, logits, Vn, 3);
        k_logsoftmax<<<Bn, 256, 0, stream>>>(logits, length, logp_out, txt_out, t);
    }
}
